// CGMPredictor_11974368821822
// MI455X (gfx1250) — matrix operand order repaired and checked by reading, not compiled, not run on hardware
//
#include <hip/hip_runtime.h>
#include <hip/hip_bf16.h>

typedef __attribute__((ext_vector_type(16))) _Float16 v16h;
typedef __attribute__((ext_vector_type(8)))  _Float16 v8h;
typedef __attribute__((ext_vector_type(2)))  _Float16 v2h;
typedef __attribute__((ext_vector_type(8)))  float    v8f;
typedef __attribute__((ext_vector_type(4)))  float    v4f;

#define NEC 512   // model width
#define HDC 64    // head dim

// permuted k-position so an A-fragment is one contiguous v16h per lane half:
// lo half reads pos 0..15  (k = 0..7,16..23), hi half reads pos 16..31 (k = 8..15,24..31)
__device__ __forceinline__ int kperm(int k) {
  return (k & 7) + ((k >> 3) & 1) * 16 + ((k >> 4) & 1) * 8;
}

// ---------------------------------------------------------------- reductions
__device__ __forceinline__ float half16_red_sum(float v) {
#pragma unroll
  for (int m = 8; m >= 1; m >>= 1) v += __shfl_xor(v, m, 32);
  return v;
}
__device__ __forceinline__ float half16_red_max(float v) {
#pragma unroll
  for (int m = 8; m >= 1; m >>= 1) v = fmaxf(v, __shfl_xor(v, m, 32));
  return v;
}

// ---------------------------------------------------------------- WMMA GEMM
// C[M,N] = act(A[M,K] @ W[K,N] + bias) (+ R)   fp32 memory, f16 WMMA compute.
// 64x64 block tile, 4 waves; LDS tiles pre-swizzled to WMMA fragment layouts;
// double-buffered software pipeline (one barrier per K-step, global-load
// latency hidden under the WMMA block of the previous step).
__global__ __launch_bounds__(128) void gemm_wmma_kernel(
    const float* __restrict__ A, const float* __restrict__ W,
    const float* __restrict__ bias, const float* R,
    float* C, int M, int N, int K, int act)
{
  __shared__ _Float16 abuf[2][64 * 32];   // [row][kperm(k)]
  __shared__ _Float16 wbuf[2][64 * 32];   // [n][k]  (transposed W tile)
  const int wave = threadIdx.x >> 5;
  const int lane = threadIdx.x & 31;
  const int hi   = lane >> 4;
  const int nl   = lane & 15;
  const int tileN = blockIdx.x << 6;
  const int tileM = blockIdx.y << 6;
  const int bKb = hi ? 16 : 0;

  // staging coordinates
  const int ar  = threadIdx.x >> 1;          // 0..63  A row
  const int ac  = (threadIdx.x & 1) << 4;    // 0|16   A k base
  const int aD0 = ar * 32 + kperm(ac);
  const int aD1 = ar * 32 + kperm(ac + 8);
  const int wkp = threadIdx.x >> 3;          // 0..15  W k-pair (rows 2kp, 2kp+1)
  const int wcB = (threadIdx.x & 7) << 3;    // 0..56  W col base (8 cols)

  v8f vz = {};
  v8f acc[4];
#pragma unroll
  for (int j = 0; j < 4; ++j) acc[j] = vz;

  v4f a0, a1, a2, a3, w00, w01, w10, w11;   // in-flight tile registers

  auto load_tile = [&](int k0) {
    const float* ap = &A[(size_t)(tileM + ar) * K + k0 + ac];
    a0 = *(const v4f*)(ap);
    a1 = *(const v4f*)(ap + 4);
    a2 = *(const v4f*)(ap + 8);
    a3 = *(const v4f*)(ap + 12);
    const float* wp0 = &W[(size_t)(k0 + 2 * wkp) * N + tileN + wcB];
    const float* wp1 = wp0 + N;
    w00 = *(const v4f*)(wp0); w01 = *(const v4f*)(wp0 + 4);
    w10 = *(const v4f*)(wp1); w11 = *(const v4f*)(wp1 + 4);
  };
  auto store_tile = [&](int bsel) {
    v8h g0, g1;
#pragma unroll
    for (int i = 0; i < 4; ++i) {
      g0[i] = (_Float16)a0[i]; g0[4 + i] = (_Float16)a1[i];
      g1[i] = (_Float16)a2[i]; g1[4 + i] = (_Float16)a3[i];
    }
    *(v8h*)&abuf[bsel][aD0] = g0;
    *(v8h*)&abuf[bsel][aD1] = g1;
#pragma unroll
    for (int i = 0; i < 4; ++i) {
      v2h p0; p0[0] = (_Float16)w00[i]; p0[1] = (_Float16)w10[i];
      *(v2h*)&wbuf[bsel][(wcB + i) * 32 + 2 * wkp] = p0;
      v2h p1; p1[0] = (_Float16)w01[i]; p1[1] = (_Float16)w11[i];
      *(v2h*)&wbuf[bsel][(wcB + 4 + i) * 32 + 2 * wkp] = p1;
    }
  };
  auto compute_tile = [&](int bsel) {
    v16h af = *(const v16h*)&abuf[bsel][((wave << 4) + nl) * 32 + (hi ? 16 : 0)];
#pragma unroll
    for (int j = 0; j < 4; ++j) {
      const _Float16* wrow = &wbuf[bsel][((j << 4) + nl) * 32 + (hi ? 8 : 0)];
      v8h b0 = *(const v8h*)wrow, b1 = *(const v8h*)(wrow + 16);
      v16h bf = __builtin_shufflevector(b0, b1, 0, 1, 2, 3, 4, 5, 6, 7, 8, 9, 10, 11, 12, 13, 14, 15);
      acc[j] = __builtin_amdgcn_wmma_f32_16x16x32_f16(false, af, false, bf,
                                                      (short)0, acc[j], false, false);
    }
  };

  const int K32 = K & ~31;
  const int nIter = K32 >> 5;
  if (nIter > 0) {
    load_tile(0);
    store_tile(0);
    for (int i = 0; i < nIter; ++i) {
      if (i + 1 < nIter) load_tile((i + 1) << 5);
      __syncthreads();
      compute_tile(i & 1);
      if (i + 1 < nIter) store_tile((i + 1) & 1);
    }
  }
  if (K32 < K) {   // tail: only the K=11 encoder GEMM reaches this
    const int k0 = K32;
    const int wr = threadIdx.x >> 2;          // 0..31 k row
    const int wc = (threadIdx.x & 3) << 4;    // 0..48 col base
    __syncthreads();
#pragma unroll
    for (int q = 0; q < 16; ++q) {
      int ka = k0 + ac + q;
      abuf[0][ar * 32 + kperm(ac + q)] =
          (_Float16)((ka < K) ? A[(size_t)(tileM + ar) * K + ka] : 0.0f);
      int kw = k0 + wr;
      wbuf[0][(wc + q) * 32 + wr] =
          (_Float16)((kw < K) ? W[(size_t)kw * N + tileN + wc + q] : 0.0f);
    }
    __syncthreads();
    compute_tile(0);
  }
  // epilogue
#pragma unroll
  for (int j = 0; j < 4; ++j)
#pragma unroll
    for (int r = 0; r < 8; ++r) {
      int mm = tileM + (wave << 4) + r + (hi ? 8 : 0);
      int nn = tileN + (j << 4) + nl;
      float v = acc[j][r];
      if (bias) v += bias[nn];
      if (act == 1)      v = fmaxf(v, 0.0f);
      else if (act == 2) v = 0.5f * v * (1.0f + erff(v * 0.70710678118654752f));
      if (R) v += R[(size_t)mm * N + nn];
      C[(size_t)mm * N + nn] = v;
    }
}

// ---------------------------------------------------------------- attention
// Q:(B,Tq,512) rope'd; KV:(B,Ty,1024) [K rope'd | V]; O:(B,Tq,512)
// grid = (Tq/64, NH, B), block = 128 (4 waves, one 16-query tile each).
// K tile [key][dim], V tile [dim][key], double-buffered; P round-trips LDS
// in the permuted-k A layout.
__global__ __launch_bounds__(128) void attn_wmma_kernel(
    const float* __restrict__ Q, const float* __restrict__ KV,
    const float* __restrict__ qt, const float* __restrict__ kt,
    float* __restrict__ O, int Tq, int Ty, float minDist)
{
  __shared__ _Float16 kbuf[2][32 * 64];     // [key][dim]
  __shared__ _Float16 vbuf[2][64 * 32];     // [dim][key]
  __shared__ _Float16 pbuf[4][16 * 32];     // [row][kperm(col)]
  const int wave = threadIdx.x >> 5;
  const int lane = threadIdx.x & 31;
  const int hi   = lane >> 4;
  const int nl   = lane & 15;
  const int h = blockIdx.y;
  const int b = blockIdx.z;
  const int qtile = (blockIdx.x << 2) + wave;
  const int aKb = hi ? 16 : 0;
  const int bKb = hi ? 16 : 0;

  // Q fragments (A layout), vectorized loads, pre-scaled by 1/sqrt(hd)
  const int qrow = b * Tq + (qtile << 4) + nl;
  const float* qp = &Q[(size_t)qrow * NEC + h * HDC + aKb];
  v16h aq0, aq1;
#pragma unroll
  for (int g = 0; g < 2; ++g)
#pragma unroll
    for (int q = 0; q < 2; ++q) {
      v4f v0 = *(const v4f*)(qp + g * 8 + q * 4);
      v4f v1 = *(const v4f*)(qp + 32 + g * 8 + q * 4);
#pragma unroll
      for (int i = 0; i < 4; ++i) {
        aq0[g * 8 + q * 4 + i] = (_Float16)(v0[i] * 0.125f);
        aq1[g * 8 + q * 4 + i] = (_Float16)(v1[i] * 0.125f);
      }
    }
  float qtimes[8];
#pragma unroll
  for (int r = 0; r < 8; ++r)
    qtimes[r] = qt[b * Tq + (qtile << 4) + r + (hi ? 8 : 0)] - minDist;

  float mrow[8], lrow[8];
  v8f vz = {};
  v8f oacc[4];
#pragma unroll
  for (int r = 0; r < 8; ++r) { mrow[r] = -1e30f; lrow[r] = 0.0f; }
#pragma unroll
  for (int t = 0; t < 4; ++t) oacc[t] = vz;

  const int keyL = threadIdx.x >> 2;        // 0..31  (K staging)
  const int dB   = (threadIdx.x & 3) << 4;  // 0..48
  const int vkp  = threadIdx.x >> 3;        // 0..15  (V staging key pair)
  const int vdB  = (threadIdx.x & 7) << 3;  // 0..56
  const int pc0  = kperm(nl);               // P store positions
  const int pc1  = kperm(16 + nl);

  v4f k0v, k1v, k2v, k3v, v00, v01, v10, v11;   // in-flight tile registers

  auto load_kv = [&](int kb0) {
    const float* kp = &KV[(size_t)(b * Ty + kb0 + keyL) * 1024 + h * HDC + dB];
    k0v = *(const v4f*)(kp);
    k1v = *(const v4f*)(kp + 4);
    k2v = *(const v4f*)(kp + 8);
    k3v = *(const v4f*)(kp + 12);
    const float* vp0 = &KV[(size_t)(b * Ty + kb0 + 2 * vkp) * 1024 + NEC + h * HDC + vdB];
    const float* vp1 = vp0 + 1024;
    v00 = *(const v4f*)(vp0); v01 = *(const v4f*)(vp0 + 4);
    v10 = *(const v4f*)(vp1); v11 = *(const v4f*)(vp1 + 4);
  };
  auto store_kv = [&](int bsel) {
    v8h g0, g1;
#pragma unroll
    for (int i = 0; i < 4; ++i) {
      g0[i] = (_Float16)k0v[i]; g0[4 + i] = (_Float16)k1v[i];
      g1[i] = (_Float16)k2v[i]; g1[4 + i] = (_Float16)k3v[i];
    }
    *(v8h*)&kbuf[bsel][keyL * 64 + dB]     = g0;
    *(v8h*)&kbuf[bsel][keyL * 64 + dB + 8] = g1;
#pragma unroll
    for (int i = 0; i < 4; ++i) {
      v2h p0; p0[0] = (_Float16)v00[i]; p0[1] = (_Float16)v10[i];
      *(v2h*)&vbuf[bsel][(vdB + i) * 32 + 2 * vkp] = p0;
      v2h p1; p1[0] = (_Float16)v01[i]; p1[1] = (_Float16)v11[i];
      *(v2h*)&vbuf[bsel][(vdB + 4 + i) * 32 + 2 * vkp] = p1;
    }
  };

  const int nIter = Ty >> 5;
  load_kv(0);
  store_kv(0);
  for (int it = 0; it < nIter; ++it) {
    const int kb0 = it << 5;
    const int bsel = it & 1;
    if (it + 1 < nIter) load_kv((it + 1) << 5);
    __syncthreads();

    v8f s[2];
#pragma unroll
    for (int st = 0; st < 2; ++st) {
      v16h bk0 = *(const v16h*)&kbuf[bsel][((st << 4) + nl) * 64 + bKb];
      v16h bk1 = *(const v16h*)&kbuf[bsel][((st << 4) + nl) * 64 + 32 + bKb];
      v8f z = vz;
      z = __builtin_amdgcn_wmma_f32_16x16x32_f16(false, aq0, false, bk0, (short)0, z, false, false);
      z = __builtin_amdgcn_wmma_f32_16x16x32_f16(false, aq1, false, bk1, (short)0, z, false, false);
      float ktime = kt[b * Ty + kb0 + (st << 4) + nl];
#pragma unroll
      for (int r = 0; r < 8; ++r)
        if (qtimes[r] < ktime) z[r] = -1e30f;
      s[st] = z;
    }
    // online softmax over this 32-key slab
    float scl[8];
#pragma unroll
    for (int r = 0; r < 8; ++r) {
      float tm = half16_red_max(fmaxf(s[0][r], s[1][r]));
      float mnew = fmaxf(mrow[r], tm);
      scl[r] = __expf(mrow[r] - mnew);
      float p0 = (s[0][r] < -1e29f) ? 0.0f : __expf(s[0][r] - mnew);
      float p1 = (s[1][r] < -1e29f) ? 0.0f : __expf(s[1][r] - mnew);
      s[0][r] = p0; s[1][r] = p1;
      float ps = half16_red_sum(p0 + p1);
      lrow[r] = lrow[r] * scl[r] + ps;
      mrow[r] = mnew;
    }
#pragma unroll
    for (int t = 0; t < 4; ++t)
#pragma unroll
      for (int r = 0; r < 8; ++r) oacc[t][r] *= scl[r];

    // P: C layout -> permuted A layout via per-wave LDS tile
#pragma unroll
    for (int r = 0; r < 8; ++r) {
      pbuf[wave][(r + (hi ? 8 : 0)) * 32 + pc0] = (_Float16)s[0][r];
      pbuf[wave][(r + (hi ? 8 : 0)) * 32 + pc1] = (_Float16)s[1][r];
    }
    __syncthreads();
    v16h pa = *(const v16h*)&pbuf[wave][nl * 32 + (hi ? 16 : 0)];

    // O += P @ V
#pragma unroll
    for (int t = 0; t < 4; ++t) {
      const _Float16* vrow = &vbuf[bsel][((t << 4) + nl) * 32 + (hi ? 8 : 0)];
      v8h c0 = *(const v8h*)vrow, c1 = *(const v8h*)(vrow + 16);
      v16h vf = __builtin_shufflevector(c0, c1, 0, 1, 2, 3, 4, 5, 6, 7, 8, 9, 10, 11, 12, 13, 14, 15);
      oacc[t] = __builtin_amdgcn_wmma_f32_16x16x32_f16(false, pa, false, vf,
                                                       (short)0, oacc[t], false, false);
    }
    if (it + 1 < nIter) store_kv((it + 1) & 1);
  }
#pragma unroll
  for (int t = 0; t < 4; ++t)
#pragma unroll
    for (int r = 0; r < 8; ++r) {
      int q = (qtile << 4) + r + (hi ? 8 : 0);
      O[(size_t)(b * Tq + q) * NEC + h * HDC + (t << 4) + nl] =
          oacc[t][r] / fmaxf(lrow[r], 1e-20f);
    }
}

// ---------------------------------------------------------------- LayerNorm
__global__ __launch_bounds__(128) void ln_kernel(
    const float* __restrict__ in, const float* __restrict__ g,
    const float* __restrict__ bt, float* __restrict__ out,
    int inT, int outT, int outOff)
{
  __shared__ float sm[4];
  const int row = blockIdx.x;
  const int b = row / inT, t = row - b * inT;
  const float* x = in + (size_t)row * NEC;
  float* y = out + (size_t)(b * outT + outOff + t) * NEC;
  float v[4]; float s = 0.0f;
#pragma unroll
  for (int i = 0; i < 4; ++i) { v[i] = x[threadIdx.x + i * 128]; s += v[i]; }
#pragma unroll
  for (int m = 16; m >= 1; m >>= 1) s += __shfl_xor(s, m, 32);
  if ((threadIdx.x & 31) == 0) sm[threadIdx.x >> 5] = s;
  __syncthreads();
  float mu = (sm[0] + sm[1] + sm[2] + sm[3]) * (1.0f / NEC);
  __syncthreads();
  float s2 = 0.0f;
#pragma unroll
  for (int i = 0; i < 4; ++i) { float d = v[i] - mu; s2 += d * d; }
#pragma unroll
  for (int m = 16; m >= 1; m >>= 1) s2 += __shfl_xor(s2, m, 32);
  if ((threadIdx.x & 31) == 0) sm[threadIdx.x >> 5] = s2;
  __syncthreads();
  float var = (sm[0] + sm[1] + sm[2] + sm[3]) * (1.0f / NEC);
  float inv = rsqrtf(var + 1e-5f);
#pragma unroll
  for (int i = 0; i < 4; ++i) {
    int c2 = threadIdx.x + i * 128;
    y[c2] = (v[i] - mu) * inv * g[c2] + bt[c2];
  }
}

// ---------------------------------------------------------------- RoPE
__global__ void rope_kernel(float* __restrict__ buf, const float* __restrict__ tp,
                            const float* __restrict__ freqs, int T, int rowC,
                            int colOff, int rows)
{
  int idx = blockIdx.x * blockDim.x + threadIdx.x;
  const int ppr = NEC / 2;
  if (idx >= rows * ppr) return;
  int row = idx / ppr;
  int p = idx - row * ppr;
  int b = row / T, t = row - b * T;
  float f = tp[b * T + t] * freqs[p & 31];
  float c = cosf(f), s = sinf(f);
  float* q = buf + (size_t)row * rowC + colOff + 2 * p;
  float a0 = q[0], a1 = q[1];
  q[0] = a0 * c - a1 * s;
  q[1] = a1 * c + a0 * s;
}

// ---------------------------------------------------------------- encoder feat
__global__ void feat_kernel(const float* __restrict__ x, const float* __restrict__ tp,
                            float scale, float shift, float* __restrict__ feat,
                            int T, int rows)
{
  int idx = blockIdx.x * blockDim.x + threadIdx.x;
  if (idx >= rows) return;
  int b = idx / T, t = idx - b * T;
  float wt[6];
  float* f = feat + (size_t)idx * 11;
#pragma unroll
  for (int d = 0; d < 6; ++d) {
    int src = t + d - 5; if (src < 0) src = 0;
    f[d] = (x[b * T + src] - shift) * scale;
    wt[d] = tp[b * T + src];
  }
#pragma unroll
  for (int j = 0; j < 5; ++j) f[6 + j] = wt[j] - wt[5];
}

__global__ void concat_t_kernel(const float* __restrict__ a,
                                const float* __restrict__ b2, float* __restrict__ out)
{
  int idx = blockIdx.x * blockDim.x + threadIdx.x;
  if (idx >= 8 * 512) return;
  int b = idx >> 9, t = idx & 511;
  out[idx] = (t < 256) ? a[b * 256 + t] : b2[b * 256 + t - 256];
}

__global__ __launch_bounds__(128) void head_kernel(
    const float* __restrict__ cls, const float* __restrict__ hw,
    const float* __restrict__ hb, float* __restrict__ out, int rows)
{
  int wave = threadIdx.x >> 5, lane = threadIdx.x & 31;
  int row = blockIdx.x * 4 + wave;
  if (row >= rows) return;
  float acc = 0.0f;
  for (int c = lane; c < NEC; c += 32) acc += cls[(size_t)row * NEC + c] * hw[c];
#pragma unroll
  for (int m = 16; m >= 1; m >>= 1) acc += __shfl_xor(acc, m, 32);
  if (lane == 0) out[row] = acc + hb[0];
}

// ================================================================ host side
namespace {

constexpr int Bb = 8, Tc = 512, Tbas = 256, Tq = 128, Tins = 512;

struct AttnP { const float *wq,*bq,*wkv,*bkv,*wo,*bo,*freqs; };
struct MlpP  { const float *wfc,*bfc,*wproj,*bproj; };
struct SaP   { const float *l1w,*l1b; AttnP at; const float *l2w,*l2b; MlpP mlp; };
struct BidirP{ const float *xl1w,*xl1b,*yl1w,*yl1b; AttnP xat, yat;
               const float *xl2w,*xl2b,*yl2w,*yl2b; MlpP xm, ym; };
struct CrossP{ const float *xl3w,*xl3b,*yl1w,*yl1b; AttnP at;
               const float *yl2w,*yl2b; MlpP mlp; };
struct EncP  { const float *w1,*b1,*w2,*b2,*lnw,*lnb; };
struct BlockP{ SaP ins_sa, cgm_sa; BidirP bid; CrossP cins, ccgm; };

struct Cursor {
  void* const* a; int i;
  const float* f() { return (const float*)a[i++]; }
};
AttnP pAttn(Cursor& c){ AttnP p; p.wq=c.f(); p.bq=c.f(); p.wkv=c.f(); p.bkv=c.f();
                        p.wo=c.f(); p.bo=c.f(); p.freqs=c.f(); return p; }
MlpP  pMlp (Cursor& c){ MlpP p; p.wfc=c.f(); p.bfc=c.f(); p.wproj=c.f(); p.bproj=c.f(); return p; }
SaP   pSa  (Cursor& c){ SaP p; p.l1w=c.f(); p.l1b=c.f(); p.at=pAttn(c);
                        p.l2w=c.f(); p.l2b=c.f(); p.mlp=pMlp(c); return p; }
BidirP pBidir(Cursor& c){ BidirP p; p.xl1w=c.f(); p.xl1b=c.f(); p.yl1w=c.f(); p.yl1b=c.f();
                          p.xat=pAttn(c); p.yat=pAttn(c);
                          p.xl2w=c.f(); p.xl2b=c.f(); p.yl2w=c.f(); p.yl2b=c.f();
                          p.xm=pMlp(c); p.ym=pMlp(c); return p; }
CrossP pCross(Cursor& c){ CrossP p; p.xl3w=c.f(); p.xl3b=c.f(); p.yl1w=c.f(); p.yl1b=c.f();
                          p.at=pAttn(c); p.yl2w=c.f(); p.yl2b=c.f(); p.mlp=pMlp(c); return p; }
EncP  pEnc (Cursor& c){ EncP p; p.w1=c.f(); p.b1=c.f(); p.w2=c.f(); p.b2=c.f();
                        p.lnw=c.f(); p.lnb=c.f(); return p; }
BlockP pBlock(Cursor& c){ BlockP p; p.ins_sa=pSa(c); p.cgm_sa=pSa(c); p.bid=pBidir(c);
                          p.cins=pCross(c); p.ccgm=pCross(c); return p; }

struct Ws { float *cgm,*ins,*cls,*ins_t,*xl,*yl,*qb,*kvb,*ob,*tmp,*h1,*feat; };

void gemm(hipStream_t s, const float* A, const float* W, const float* bias,
          const float* R, float* C, int M, int N, int K, int act)
{
  dim3 g(N / 64, M / 64), bl(128);
  gemm_wmma_kernel<<<g, bl, 0, s>>>(A, W, bias, R, C, M, N, K, act);
}
void ln(hipStream_t s, const float* in, const float* g, const float* b, float* out,
        int rows, int inT, int outT, int outOff)
{
  ln_kernel<<<dim3(rows), dim3(128), 0, s>>>(in, g, b, out, inT, outT, outOff);
}
void rope(hipStream_t s, float* buf, const float* tp, const float* fr, int T,
          int rowC, int colOff)
{
  int rows = Bb * T;
  int total = rows * (NEC / 2);
  rope_kernel<<<dim3((total + 255) / 256), dim3(256), 0, s>>>(buf, tp, fr, T, rowC, colOff, rows);
}
void attn_layer(hipStream_t s, const AttnP& p, const Ws& w,
                const float* qsrc, const float* qtimes, int Tx,
                const float* ksrc, const float* ktimes, int Ty, float* xio)
{
  int Mq = Bb * Tx, Mk = Bb * Ty;
  gemm(s, qsrc, p.wq, p.bq, nullptr, w.qb, Mq, 512, 512, 0);
  gemm(s, ksrc, p.wkv, p.bkv, nullptr, w.kvb, Mk, 1024, 512, 0);
  rope(s, w.qb, qtimes, p.freqs, Tx, 512, 0);
  rope(s, w.kvb, ktimes, p.freqs, Ty, 1024, 0);
  attn_wmma_kernel<<<dim3(Tx / 64, 8, Bb), dim3(128), 0, s>>>(
      w.qb, w.kvb, qtimes, ktimes, w.ob, Tx, Ty, 0.0f);
  gemm(s, w.ob, p.wo, p.bo, xio, xio, Mq, 512, 512, 0);
}
void mlp_layer(hipStream_t s, const MlpP& p, const Ws& w,
               const float* lnw, const float* lnb, float* xio, int Tx)
{
  int M = Bb * Tx;
  ln(s, xio, lnw, lnb, w.xl, M, Tx, Tx, 0);
  gemm(s, w.xl, p.wfc, p.bfc, nullptr, w.tmp, M, 512, 512, 2);
  gemm(s, w.tmp, p.wproj, p.bproj, xio, xio, M, 512, 512, 0);
}
void sa_layer(hipStream_t s, const SaP& p, const Ws& w, float* x, const float* xt, int Tx)
{
  ln(s, x, p.l1w, p.l1b, w.xl, Bb * Tx, Tx, Tx, 0);
  attn_layer(s, p.at, w, w.xl, xt, Tx, w.xl, xt, Tx, x);
  mlp_layer(s, p.mlp, w, p.l2w, p.l2b, x, Tx);
}
void bidir_layer(hipStream_t s, const BidirP& p, const Ws& w,
                 float* cgm, const float* cgmT, float* ins, const float* insT)
{
  ln(s, ins, p.xl1w, p.xl1b, w.xl, Bb * Tins, Tins, Tins, 0);
  ln(s, cgm, p.yl1w, p.yl1b, w.yl, Bb * Tc, Tc, Tc, 0);
  attn_layer(s, p.xat, w, w.xl, insT, Tins, w.yl, cgmT, Tc, ins);
  attn_layer(s, p.yat, w, w.yl, cgmT, Tc, w.xl, insT, Tins, cgm);
  mlp_layer(s, p.xm, w, p.xl2w, p.xl2b, ins, Tins);
  mlp_layer(s, p.ym, w, p.yl2w, p.yl2b, cgm, Tc);
}
void cross_layer(hipStream_t s, const CrossP& p, const Ws& w,
                 float* cls, const float* clsT, const float* x, const float* xT, int Tx)
{
  ln(s, cls, p.yl1w, p.yl1b, w.xl, Bb * Tq, Tq, Tq, 0);
  ln(s, x, p.xl3w, p.xl3b, w.yl, Bb * Tx, Tx, Tx, 0);
  attn_layer(s, p.at, w, w.xl, clsT, Tq, w.yl, xT, Tx, cls);
  mlp_layer(s, p.mlp, w, p.yl2w, p.yl2b, cls, Tq);
}
void enc_layer(hipStream_t s, const EncP& p, const Ws& w,
               const float* x, const float* tp, float scale, float shift,
               int T, float* out, int outT, int outOff)
{
  int rows = Bb * T;
  feat_kernel<<<dim3((rows + 255) / 256), dim3(256), 0, s>>>(x, tp, scale, shift, w.feat, T, rows);
  gemm(s, w.feat, p.w1, p.b1, nullptr, w.h1, rows, 2048, 11, 1);
  gemm(s, w.h1, p.w2, p.b2, nullptr, w.tmp, rows, 512, 2048, 1);
  ln(s, w.tmp, p.lnw, p.lnb, out, rows, T, outT, outOff);
}

} // namespace

extern "C" void kernel_launch(void* const* d_in, const int* in_sizes, int n_in,
                              void* d_out, int out_size, void* d_ws, size_t ws_size,
                              hipStream_t stream)
{
  (void)in_sizes; (void)n_in; (void)out_size; (void)ws_size;
  Cursor c{d_in, 0};
  const float* inp_cgm = c.f();  const float* inp_basal = c.f(); const float* inp_bolus = c.f();
  const float* cgm_t  = c.f();   const float* basal_t = c.f();   const float* bolus_t = c.f();
  const float* cls_t  = c.f();
  EncP cgmE = pEnc(c), basalE = pEnc(c), bolusE = pEnc(c);
  BlockP blocks[4];
  for (int i = 0; i < 4; ++i) blocks[i] = pBlock(c);
  const float* ln_w = c.f(); (void)ln_w;   // unused by reference forward
  const float* head_w = c.f(); const float* head_b = c.f();

  size_t off = 0;
  auto alloc = [&](size_t n) {
    float* p = (float*)((char*)d_ws + off);
    off += ((n * sizeof(float) + 255) & ~(size_t)255);
    return p;
  };
  Ws w;
  w.cgm  = alloc((size_t)Bb * Tc * 512);
  w.ins  = alloc((size_t)Bb * Tins * 512);
  w.cls  = alloc((size_t)Bb * Tq * 512);
  w.ins_t= alloc((size_t)Bb * Tins);
  w.xl   = alloc((size_t)Bb * 512 * 512);
  w.yl   = alloc((size_t)Bb * 512 * 512);
  w.qb   = alloc((size_t)Bb * 512 * 512);
  w.kvb  = alloc((size_t)Bb * 512 * 1024);
  w.ob   = alloc((size_t)Bb * 512 * 512);
  w.tmp  = alloc((size_t)Bb * 512 * 512);
  w.h1   = alloc((size_t)Bb * 512 * 2048);
  w.feat = alloc((size_t)Bb * 512 * 11);

  // encoders
  enc_layer(stream, cgmE,   w, inp_cgm,   cgm_t,   1.0f / 50.0f, 150.0f, Tc,   w.cgm, Tc,   0);
  enc_layer(stream, basalE, w, inp_basal, basal_t, 1.0f / 1.5f,  0.0f,   Tbas, w.ins, Tins, 0);
  enc_layer(stream, bolusE, w, inp_bolus, bolus_t, 1.0f / 2.0f,  0.0f,   Tbas, w.ins, Tins, 256);
  concat_t_kernel<<<dim3((Bb * Tins + 255) / 256), dim3(256), 0, stream>>>(basal_t, bolus_t, w.ins_t);
  hipMemsetAsync(w.cls, 0, (size_t)Bb * Tq * 512 * sizeof(float), stream);

  // transformer blocks
  for (int i = 0; i < 4; ++i) {
    const BlockP& bp = blocks[i];
    sa_layer(stream, bp.ins_sa, w, w.ins, w.ins_t, Tins);
    sa_layer(stream, bp.cgm_sa, w, w.cgm, cgm_t, Tc);
    bidir_layer(stream, bp.bid, w, w.cgm, cgm_t, w.ins, w.ins_t);
    cross_layer(stream, bp.cins, w, w.cls, cls_t, w.ins, w.ins_t, Tins);
    cross_layer(stream, bp.ccgm, w, w.cls, cls_t, w.cgm, cgm_t, Tc);
  }

  head_kernel<<<dim3((Bb * Tq + 3) / 4), dim3(128), 0, stream>>>(
      w.cls, head_w, head_b, (float*)d_out, Bb * Tq);
}
